// SR_GNN_17978733101798
// MI455X (gfx1250) — matrix operand order repaired and checked by reading, not compiled, not run on hardware
//
#include <hip/hip_runtime.h>

// SR-GNN forward for MI455X (gfx1250, wave32).
// H=128 fixed; N,E,V derived from in_sizes; S=2048 (reference constant).

#define HID 128
#define S_SESS 2048

typedef __attribute__((ext_vector_type(16))) _Float16 v16h;
typedef __attribute__((ext_vector_type(8)))  _Float16 v8h;
typedef __attribute__((ext_vector_type(8)))  float    v8f;

// ---------------- WMMA fragment loaders (ISA 7.12.2, wave32) ----------------
// A: 16x32 f16 (MxK), row-major source with leading dim lda.
// lanes 0-15  : M=lane,    K = k0+0..15
// lanes 16-31 : M=lane-16, K = k0+16..31
__device__ __forceinline__ v16h load_a_frag(const _Float16* __restrict__ A, int lda,
                                            int m0, int k0, int lane) {
  const int m  = m0 + (lane & 15);
  const int kb = k0 + ((lane >> 4) << 4);
  const _Float16* p = A + (size_t)m * lda + kb;
  v8h lo = *(const v8h*)p;
  v8h hi = *(const v8h*)(p + 8);
  v16h r;
#pragma unroll
  for (int i = 0; i < 8; ++i) { r[i] = lo[i]; r[8 + i] = hi[i]; }
  return r;
}

// B: 32x16 f16 (KxN); source is Bt[N][K] row-major (i.e. B transposed), so
// lane n loads 16 *contiguous* K values of output column n:
// lanes 0-15 : N=lane,    K = k0+0..15
// lanes16-31 : N=lane-16, K = k0+16..31
__device__ __forceinline__ v16h load_b_frag(const _Float16* __restrict__ Bt, int ldb,
                                            int n0, int k0, int lane) {
  const int n  = n0 + (lane & 15);
  const int kb = k0 + ((lane >> 4) << 4);
  return *(const v16h*)(Bt + (size_t)n * ldb + kb);
}

// ---------------- generic NT-form WMMA GEMM ----------------
// C[M,Ncols] (f32) = A[M,K](f16,row-major) x Bt[Ncols,K](f16,row-major)^T + bias
// Each wave computes a 16 x (16*NT) strip.
// REQUIREMENTS: M%16==0, K%32==0, (Ncols/16)%NT==0 (no edge guards -> no EXEC
// divergence around WMMA; tile indices are forced into SGPRs via readfirstlane
// so all control flow is scalar and EXEC stays all-ones as the ISA requires).
template <int NT, bool NTSTORE>
__global__ void gemm_nt_wmma(const _Float16* __restrict__ A,
                             const _Float16* __restrict__ Bt,
                             const float* __restrict__ bias,
                             float* __restrict__ C,
                             _Float16* __restrict__ Ch,
                             int M, int Ncols, int K) {
  const int lane = threadIdx.x & 31;
  // wave id is uniform across the wave; readfirstlane makes the compiler see it
  // as an SGPR so the guards below become scalar branches (no EXEC masking).
  const int wave = __builtin_amdgcn_readfirstlane(threadIdx.x >> 5);
  const int ngrp = (Ncols >> 4) / NT;                       // exact
  const int tile = blockIdx.x * (blockDim.x >> 5) + wave;   // scalar
  if (tile >= (M >> 4) * ngrp) return;                      // scalar branch
  const int m0 = (tile / ngrp) << 4;
  const int n0 = (tile % ngrp) * (NT << 4);

  v8f acc[NT] = {};
  for (int k0 = 0; k0 < K; k0 += 32) {
    v16h a = load_a_frag(A, K, m0, k0, lane);
#pragma unroll
    for (int t = 0; t < NT; ++t) {
      v16h b = load_b_frag(Bt, K, n0 + (t << 4), k0, lane);
      acc[t] = __builtin_amdgcn_wmma_f32_16x16x32_f16(
          false, a, false, b, (short)0, acc[t], false, false);
    }
  }

  // C/D layout: VGPR r, lanes 0-15 -> M=m0+r, lanes 16-31 -> M=m0+8+r; N=n0+lane%16
  const int mbase = m0 + ((lane >> 4) << 3);
#pragma unroll
  for (int t = 0; t < NT; ++t) {
    const int n  = n0 + (t << 4) + (lane & 15);
    const float bv = bias ? bias[n] : 0.0f;
#pragma unroll
    for (int r = 0; r < 8; ++r) {
      const size_t off = (size_t)(mbase + r) * (size_t)Ncols + n;
      const float val = acc[t][r] + bv;
      if (NTSTORE) __builtin_nontemporal_store(val, C + off);
      else         C[off] = val;
      if (Ch)      Ch[off] = (_Float16)val;
    }
  }
}

// ---------------- small helper kernels ----------------
__global__ void f32_to_f16_kernel(const float* __restrict__ s, _Float16* __restrict__ d, int n) {
  int i = blockIdx.x * blockDim.x + threadIdx.x;
  if (i < n) d[i] = (_Float16)s[i];
}

__global__ void zero_f32_kernel(float* __restrict__ p, int n) {
  int i = blockIdx.x * blockDim.x + threadIdx.x;
  if (i < n) p[i] = 0.0f;
}

// ggc_wt[j][k] = (f16) ggc_w[k][j]   (128x128)
__global__ void transpose_ggc_kernel(const float* __restrict__ s, _Float16* __restrict__ d) {
  int j = blockIdx.x, k = threadIdx.x;
  d[j * HID + k] = (_Float16)s[k * HID + j];
}

// h0 = embed[x]  (f32 + f16 copies)
__global__ void gather_embed_kernel(const int* __restrict__ x, const float* __restrict__ embed,
                                    float* __restrict__ h0, _Float16* __restrict__ h0h, int nTot) {
  int t = blockIdx.x * blockDim.x + threadIdx.x;
  if (t >= nTot) return;
  int i = t >> 7, h = t & 127;
  float v = embed[(size_t)x[i] * HID + h];
  h0[t] = v; h0h[t] = (_Float16)v;
}

// agg[dst] += w_e * m[src]
__global__ void edge_scatter_kernel(const int* __restrict__ src, const int* __restrict__ dst,
                                    const float* __restrict__ w, const float* __restrict__ m,
                                    float* __restrict__ agg, int eTot) {
  int t = blockIdx.x * blockDim.x + threadIdx.x;
  if (t >= eTot) return;
  int e = t >> 7, h = t & 127;
  atomicAdd(&agg[(size_t)dst[e] * HID + h], w[e] * m[(size_t)src[e] * HID + h]);
}

// GRU cell elementwise
__global__ void gru_kernel(const float* __restrict__ gi, const float* __restrict__ gh,
                           const float* __restrict__ h0,
                           float* __restrict__ v, _Float16* __restrict__ vh, int nTot) {
  int t = blockIdx.x * blockDim.x + threadIdx.x;
  if (t >= nTot) return;
  int i = t >> 7, h = t & 127;
  const float* gir = gi + (size_t)i * 384;
  const float* ghr = gh + (size_t)i * 384;
  float r  = 1.0f / (1.0f + __expf(-(gir[h]       + ghr[h])));
  float zg = 1.0f / (1.0f + __expf(-(gir[128 + h] + ghr[128 + h])));
  float nn = tanhf(gir[256 + h] + r * ghr[256 + h]);
  float out = (1.0f - zg) * nn + zg * h0[t];
  v[t] = out; vh[t] = (_Float16)out;
}

// last node index per session (batch sorted, sessions non-empty)
__global__ void last_idx_kernel(const int* __restrict__ batch, int* __restrict__ li, int nN) {
  int i = blockIdx.x * blockDim.x + threadIdx.x;
  if (i >= nN) return;
  if (i == nN - 1 || batch[i + 1] != batch[i]) li[batch[i]] = i;
}

__global__ void gather_last_kernel(const float* __restrict__ v, const int* __restrict__ li,
                                   float* __restrict__ s_l, _Float16* __restrict__ s_lh, int sTot) {
  int t = blockIdx.x * blockDim.x + threadIdx.x;
  if (t >= sTot) return;
  int s = t >> 7, h = t & 127;
  float val = v[(size_t)li[s] * HID + h];
  s_l[t] = val; s_lh[t] = (_Float16)val;
}

// one wave32 per node: alpha = q . sigmoid(u1[batch]+u2[node]) + qb ; s_g[batch] += alpha*v
__global__ void alpha_scatter_kernel(const float* __restrict__ u1, const float* __restrict__ u2,
                                     const float* __restrict__ qw, const float* __restrict__ qb,
                                     const int* __restrict__ batch, const float* __restrict__ v,
                                     float* __restrict__ s_g, int nN) {
  int node = blockIdx.x * (blockDim.x >> 5) + (threadIdx.x >> 5);
  if (node >= nN) return;
  int lane = threadIdx.x & 31;
  int b = batch[node];
  float sum = 0.0f;
#pragma unroll
  for (int c = 0; c < 4; ++c) {
    int h = lane + 32 * c;
    float s = 1.0f / (1.0f + __expf(-(u1[(size_t)b * HID + h] + u2[(size_t)node * HID + h])));
    sum += qw[h] * s;
  }
#pragma unroll
  for (int off = 16; off; off >>= 1) sum += __shfl_xor(sum, off, 32);
  float alpha = sum + qb[0];
#pragma unroll
  for (int c = 0; c < 4; ++c) {
    int h = lane + 32 * c;
    atomicAdd(&s_g[(size_t)b * HID + h], alpha * v[(size_t)node * HID + h]);
  }
}

// cat = [s_l | s_g] as f16, row stride 256
__global__ void build_cat_kernel(const float* __restrict__ s_l, const float* __restrict__ s_g,
                                 _Float16* __restrict__ cat, int sTot) {
  int t = blockIdx.x * blockDim.x + threadIdx.x;
  if (t >= sTot) return;
  int s = t >> 7, h = t & 127;
  cat[(size_t)s * 256 + h]       = (_Float16)s_l[t];
  cat[(size_t)s * 256 + 128 + h] = (_Float16)s_g[t];
}

// ---------------- host launcher ----------------
extern "C" void kernel_launch(void* const* d_in, const int* in_sizes, int n_in,
                              void* d_out, int out_size, void* d_ws, size_t ws_size,
                              hipStream_t stream) {
  const int*   x     = (const int*)d_in[0];
  const int*   ei    = (const int*)d_in[1];   // [2,E] row-major: src=ei, dst=ei+E
  const float* ew    = (const float*)d_in[2];
  const int*   batch = (const int*)d_in[3];
  const float* embed = (const float*)d_in[5];
  const float* ggc_w = (const float*)d_in[6];
  const float* w_ih  = (const float*)d_in[7];
  const float* w_hh  = (const float*)d_in[8];
  const float* b_ih  = (const float*)d_in[9];
  const float* b_hh  = (const float*)d_in[10];
  const float* W1w   = (const float*)d_in[11];
  const float* W1b   = (const float*)d_in[12];
  const float* W2w   = (const float*)d_in[13];
  const float* W2b   = (const float*)d_in[14];
  const float* qw    = (const float*)d_in[15];
  const float* qb    = (const float*)d_in[16];
  const float* W3w   = (const float*)d_in[17];
  const float* W3b   = (const float*)d_in[18];
  float* z = (float*)d_out;

  const int N = in_sizes[0];            // 24576
  const int E = in_sizes[2];            // 49152
  const int V = in_sizes[5] / HID;      // 100000
  const int S = S_SESS;                 // 2048

  // bump allocator on d_ws
  char* wp = (char*)d_ws;
  auto alloc = [&](size_t bytes) -> void* {
    void* p = (void*)wp;
    wp += (bytes + 255) & ~(size_t)255;
    return p;
  };
  _Float16* embed_h = (_Float16*)alloc((size_t)V * HID * 2);
  _Float16* ggc_wt  = (_Float16*)alloc((size_t)HID * HID * 2);
  _Float16* wih_h   = (_Float16*)alloc((size_t)3 * HID * HID * 2);
  _Float16* whh_h   = (_Float16*)alloc((size_t)3 * HID * HID * 2);
  _Float16* W1h     = (_Float16*)alloc((size_t)HID * HID * 2);
  _Float16* W2h     = (_Float16*)alloc((size_t)HID * HID * 2);
  _Float16* W3h     = (_Float16*)alloc((size_t)HID * 2 * HID * 2);
  float*    h0      = (float*)   alloc((size_t)N * HID * 4);
  _Float16* h0h     = (_Float16*)alloc((size_t)N * HID * 2);
  float*    m       = (float*)   alloc((size_t)N * HID * 4);
  float*    agg     = (float*)   alloc((size_t)N * HID * 4);
  _Float16* agg_h   = (_Float16*)alloc((size_t)N * HID * 2);
  float*    gi      = (float*)   alloc((size_t)N * 3 * HID * 4);
  float*    gh      = (float*)   alloc((size_t)N * 3 * HID * 4);
  float*    v       = (float*)   alloc((size_t)N * HID * 4);
  _Float16* vh      = (_Float16*)alloc((size_t)N * HID * 2);
  float*    u1      = (float*)   alloc((size_t)S * HID * 4);
  float*    u2      = (float*)   alloc((size_t)N * HID * 4);
  int*      li      = (int*)     alloc((size_t)S * 4);
  float*    s_l     = (float*)   alloc((size_t)S * HID * 4);
  _Float16* s_lh    = (_Float16*)alloc((size_t)S * HID * 2);
  float*    s_g     = (float*)   alloc((size_t)S * HID * 4);
  _Float16* cat     = (_Float16*)alloc((size_t)S * 2 * HID * 2);
  float*    s_h     = (float*)   alloc((size_t)S * HID * 4);
  _Float16* s_hh    = (_Float16*)alloc((size_t)S * HID * 2);

  auto cvt = [&](const float* s, _Float16* d, int n) {
    f32_to_f16_kernel<<<(n + 255) / 256, 256, 0, stream>>>(s, d, n);
  };
  // All GEMMs use NT=2: (Ncols/16) is even for every call (8, 24, 6250),
  // so no edge handling is required inside the kernel.
  auto gemm = [&](const _Float16* A, const _Float16* Bt, const float* bias,
                  float* C, _Float16* Ch, int M, int Nc, int K, bool nts) {
    constexpr int NT = 2;
    const int ngrp   = (Nc / 16) / NT;
    const int total  = (M / 16) * ngrp;
    const int blocks = (total + 7) / 8;         // 8 waves / block (256 thr)
    if (nts) gemm_nt_wmma<NT, true ><<<blocks, 256, 0, stream>>>(A, Bt, bias, C, Ch, M, Nc, K);
    else     gemm_nt_wmma<NT, false><<<blocks, 256, 0, stream>>>(A, Bt, bias, C, Ch, M, Nc, K);
  };

  // 1) weight conversions (f32 -> f16; all already in NT orientation except ggc_w)
  cvt(embed, embed_h, V * HID);
  cvt(w_ih, wih_h, 3 * HID * HID);
  cvt(w_hh, whh_h, 3 * HID * HID);
  cvt(W1w, W1h, HID * HID);
  cvt(W2w, W2h, HID * HID);
  cvt(W3w, W3h, HID * 2 * HID);
  transpose_ggc_kernel<<<HID, HID, 0, stream>>>(ggc_w, ggc_wt);

  // 2) embedding gather
  gather_embed_kernel<<<(N * HID + 255) / 256, 256, 0, stream>>>(x, embed, h0, h0h, N * HID);

  // 3) m = h0 @ ggc_w            [N,128]
  gemm(h0h, ggc_wt, nullptr, m, nullptr, N, HID, HID, false);

  // 4) agg = segment_sum(w * m[src] -> dst)
  zero_f32_kernel<<<(N * HID + 255) / 256, 256, 0, stream>>>(agg, N * HID);
  edge_scatter_kernel<<<(E * HID + 255) / 256, 256, 0, stream>>>(ei, ei + E, ew, m, agg, E * HID);
  cvt(agg, agg_h, N * HID);

  // 5) GRU gates + cell          gi=[N,384], gh=[N,384]
  gemm(agg_h, wih_h, b_ih, gi, nullptr, N, 3 * HID, HID, false);
  gemm(h0h,   whh_h, b_hh, gh, nullptr, N, 3 * HID, HID, false);
  gru_kernel<<<(N * HID + 255) / 256, 256, 0, stream>>>(gi, gh, h0, v, vh, N * HID);

  // 6) session last node + attention pooling
  last_idx_kernel<<<(N + 255) / 256, 256, 0, stream>>>(batch, li, N);
  gather_last_kernel<<<(S * HID + 255) / 256, 256, 0, stream>>>(v, li, s_l, s_lh, S * HID);
  gemm(s_lh, W1h, W1b, u1, nullptr, S, HID, HID, false);   // u1 = s_l@W1^T + b1
  gemm(vh,   W2h, W2b, u2, nullptr, N, HID, HID, false);   // u2 = v  @W2^T + b2
  zero_f32_kernel<<<(S * HID + 255) / 256, 256, 0, stream>>>(s_g, S * HID);
  alpha_scatter_kernel<<<(N + 7) / 8, 256, 0, stream>>>(u1, u2, qw, qb, batch, v, s_g, N);

  // 7) s_h = [s_l|s_g] @ W3^T + b3      [S,128]  (also keep f16 copy)
  build_cat_kernel<<<(S * HID + 255) / 256, 256, 0, stream>>>(s_l, s_g, cat, S * HID);
  gemm(cat, W3h, W3b, s_h, s_hh, S, HID, 2 * HID, false);

  // 8) z = s_h @ embed^T         [S,V] — write-bound (819MB), embed_h L2-resident
  gemm(s_hh, embed_h, nullptr, z, nullptr, S, V, HID, true);
}
